// LSTM_81020263072363
// MI455X (gfx1250) — matrix operand order repaired and checked by reading, not compiled, not run on hardware
//
#include <hip/hip_runtime.h>
#include <hip/hip_bf16.h>
#include <math.h>

#define BATCH 64
#define SEQ   512
#define INSZ  1024
#define HID   1024
#define G4    4096            // 4*HID
#define NBLK_PERSIST 64       // persistent kernel blocks (1 hidden tile each)
#define LDSLD 1032            // padded LDS row stride (elements): 2064B = 4 banks mod 64

typedef __attribute__((ext_vector_type(16))) __bf16 v16bf;
typedef __attribute__((ext_vector_type(8)))  __bf16 v8bf;
typedef __attribute__((ext_vector_type(8)))  float  v8f;
typedef __attribute__((ext_vector_type(4)))  float  v4f;

// ---------- workspace layout (bytes) ----------
#define WIT_OFF  0u                    // Wi^T bf16 [4096][1024]  (8 MiB)
#define WHT_OFF  8388608u              // Wh^T bf16 [4096][1024]  (8 MiB)
#define BIAS_OFF 16777216u             // bi+bh fp32 [4096]
#define HBF_OFF  16793600u             // h bf16 double-buffer 2*[64][1024]
#define BAR_OFF  17055744u             // grid barrier {cnt, gen}
#define GX_OFF   33554432u             // gates_x fp32 [512][64][4096] (512 MiB)

// ================= WMMA fragment loaders (wave32, bf16 16x16x32) =========
// A (16x32) per ISA 7.12.2: lane L holds row (L&15);
//   elems 0..7  -> K = k0 + 8*(L>>4) + i ; elems 8..15 -> K = +16 more.
__device__ __forceinline__ v16bf load_a_bf16(const __bf16* __restrict__ base,
                                             int row0, int ld, int k0, int lane) {
  const __bf16* p = base + (size_t)(row0 + (lane & 15)) * ld + k0 + ((lane >> 4) << 3);
  v8bf lo = *(const v8bf*)p;
  v8bf hi = *(const v8bf*)(p + 16);
  return __builtin_shufflevector(lo, hi, 0,1,2,3,4,5,6,7,8,9,10,11,12,13,14,15);
}

// A fragment from an fp32 row-major source, converted inline to bf16.
__device__ __forceinline__ v16bf load_a_f32(const float* __restrict__ base,
                                            int row0, int ld, int k0, int lane) {
  const float* p = base + (size_t)(row0 + (lane & 15)) * ld + k0 + ((lane >> 4) << 3);
  v4f x0 = *(const v4f*)(p);
  v4f x1 = *(const v4f*)(p + 4);
  v4f x2 = *(const v4f*)(p + 16);
  v4f x3 = *(const v4f*)(p + 20);
  v16bf a;
#pragma unroll
  for (int i = 0; i < 4; ++i) {
    a[i]      = (__bf16)x0[i];
    a[i + 4]  = (__bf16)x1[i];
    a[i + 8]  = (__bf16)x2[i];
    a[i + 12] = (__bf16)x3[i];
  }
  return a;
}

// B (32x16) from pre-transposed weights Wt[n][k] (ld = K), global memory:
// lane L holds column (L&15); elems 0..7 -> K = k0 + 8*(L>>4) + i ; elems 8..15 -> K = +16 more.
__device__ __forceinline__ v16bf load_b_bf16(const __bf16* __restrict__ wt,
                                             int ncol0, int ld, int k0, int lane) {
  const __bf16* p = wt + (size_t)(ncol0 + (lane & 15)) * ld + k0 + ((lane >> 4) << 3);
  v8bf lo = *(const v8bf*)p;
  v8bf hi = *(const v8bf*)(p + 16);
  return __builtin_shufflevector(lo, hi, 0,1,2,3,4,5,6,7,8,9,10,11,12,13,14,15);
}

// B fragment from the LDS-persisted weight slice (row stride LDSLD, 16B aligned).
__device__ __forceinline__ v16bf load_b_lds(const __bf16* bsl, int row0,
                                            int k0, int lane) {
  const __bf16* p = bsl + (row0 + (lane & 15)) * LDSLD + k0 + ((lane >> 4) << 3);
  v8bf lo = *(const v8bf*)p;
  v8bf hi = *(const v8bf*)(p + 16);
  return __builtin_shufflevector(lo, hi, 0,1,2,3,4,5,6,7,8,9,10,11,12,13,14,15);
}

__device__ __forceinline__ v8f wmma_bf16(v16bf a, v16bf b, v8f c) {
  return __builtin_amdgcn_wmma_f32_16x16x32_bf16(false, a, false, b,
                                                 (short)0, c, false, false);
}

// Saturating fast gate math: v_exp_f32 + v_rcp_f32 (TRANS ops co-issue with XDL).
#define LOG2E 1.4426950408889634f
__device__ __forceinline__ float fast_sigmoid(float x) {
  float e = __builtin_amdgcn_exp2f(-LOG2E * x);          // exp(-x)
  return __builtin_amdgcn_rcpf(1.0f + e);                // +-inf saturate to 1/0
}
__device__ __forceinline__ float fast_tanh(float x) {
  return 2.0f * fast_sigmoid(2.0f * x) - 1.0f;           // saturates to +-1
}

// ================= prep: transpose+convert weights to bf16 ================
__global__ void lstm_transpose_cvt(const float* __restrict__ W,  // [K][N] fp32
                                   __bf16* __restrict__ Wt,      // [N][K] bf16
                                   int K, int N) {
  __shared__ float tile[32][33];
  int k0 = blockIdx.x * 32, n0 = blockIdx.y * 32;
  int tx = threadIdx.x, ty = threadIdx.y;        // 32 x 8
#pragma unroll
  for (int i = ty; i < 32; i += 8)
    tile[i][tx] = W[(size_t)(k0 + i) * N + n0 + tx];
  __syncthreads();
#pragma unroll
  for (int i = ty; i < 32; i += 8)
    Wt[(size_t)(n0 + i) * K + k0 + tx] = (__bf16)tile[tx][i];
}

// ================= prep: bias fuse, h buffers, barrier ====================
__global__ void lstm_prep(const float* __restrict__ bi, const float* __restrict__ bh,
                          float* __restrict__ bias, __bf16* __restrict__ hbf,
                          unsigned* __restrict__ bar) {
  int idx = blockIdx.x * blockDim.x + threadIdx.x;      // 65536 threads
  if (idx < G4) bias[idx] = bi[idx] + bh[idx];
  hbf[idx]                 = (__bf16)0.0f;              // buffer 0
  hbf[idx + BATCH * HID]   = (__bf16)0.0f;              // buffer 1
  if (idx == 0) { bar[0] = 0u; bar[1] = 0u; }
}

// ================= phase 1: gates_x = X @ Wi + b (whole-GPU GEMM) =========
// X is [B][T][IN]; flat row r = b*SEQ + t. Output laid out [t][b][4096].
// Inner loop is software-pipelined: fragments for k+1 are loaded before the
// WMMAs of k issue, so each wave keeps ~12 b128 loads in flight per 4 WMMAs.
__global__ void __launch_bounds__(256)
lstm_xgemm(const float* __restrict__ X, const __bf16* __restrict__ Wit,
           const float* __restrict__ bias, float* __restrict__ gx) {
  int lane = threadIdx.x & 31;
  int wid  = blockIdx.x * 8 + (threadIdx.x >> 5);   // 131072 waves
  int mt   = wid >> 6;                              // 0..2047 (16-row tile)
  int n0   = (wid & 63) * 64;                       // 4 N-tiles of 16

  v8f acc[4] = {v8f{}, v8f{}, v8f{}, v8f{}};

  v16bf a_cur = load_a_f32(X, mt * 16, INSZ, 0, lane);
  v16bf b_cur[4];
#pragma unroll
  for (int j = 0; j < 4; ++j) b_cur[j] = load_b_bf16(Wit, n0 + j * 16, INSZ, 0, lane);

#pragma unroll 4
  for (int kt = 0; kt < INSZ; kt += 32) {
    int kn = (kt + 32) & (INSZ - 1);                // wrap: last prefetch harmless
    v16bf a_nxt = load_a_f32(X, mt * 16, INSZ, kn, lane);
    v16bf b_nxt[4];
#pragma unroll
    for (int j = 0; j < 4; ++j) b_nxt[j] = load_b_bf16(Wit, n0 + j * 16, INSZ, kn, lane);
#pragma unroll
    for (int j = 0; j < 4; ++j) acc[j] = wmma_bf16(a_cur, b_cur[j], acc[j]);
    a_cur = a_nxt;
#pragma unroll
    for (int j = 0; j < 4; ++j) b_cur[j] = b_nxt[j];
  }

  // C/D layout: VGPR v, lanes 0-15 -> M=v, N=lane; lanes 16-31 -> M=v+8.
  int half = lane >> 4, ncol = lane & 15;
#pragma unroll
  for (int j = 0; j < 4; ++j) {
    int n = n0 + j * 16 + ncol;
    float bv = bias[n];
#pragma unroll
    for (int v = 0; v < 8; ++v) {
      int r = mt * 16 + v + 8 * half;               // r = b*SEQ + t
      int bidx = r >> 9, t = r & (SEQ - 1);
      gx[((size_t)t * BATCH + bidx) * G4 + n] = acc[j][v] + bv;
    }
  }
}

// ================= grid-wide barrier for the persistent kernel ============
__device__ __forceinline__ void grid_barrier(unsigned* cnt, unsigned* gen) {
  __threadfence();
  __syncthreads();
  if (threadIdx.x == 0) {
    volatile unsigned* vgen = (volatile unsigned*)gen;
    unsigned g = *vgen;
    if (atomicAdd(cnt, 1u) == (unsigned)NBLK_PERSIST - 1u) {
      *cnt = 0u;
      __threadfence();
      atomicAdd(gen, 1u);
    } else {
      while (*vgen == g) { __builtin_amdgcn_s_sleep(2); }
    }
  }
  __syncthreads();
  __threadfence();
}

// ================= phase 2: persistent recurrent kernel ===================
// 64 blocks x 4 waves. Block owns hidden tile jt = blockIdx.x (cols jt*16..+15
// of all four gates); its Wh^T slice (64 rows x 1024 K, 129KB) is persisted in
// LDS for all 512 steps via async global->LDS copies. Wave = batch tile mt;
// all four gate tiles accumulate in-register; cell state c lives in VGPRs for
// the entire sequence.
__global__ void __launch_bounds__(128)
lstm_recurrent(const float* __restrict__ gx, const __bf16* __restrict__ Wht,
               __bf16* __restrict__ hbf, float* __restrict__ out,
               unsigned* __restrict__ bar) {
  __shared__ __bf16 bsl[64 * LDSLD];                // 129 KiB of gfx1250's 320KB LDS

  int lane = threadIdx.x & 31;
  int mt   = threadIdx.x >> 5;                      // 0..3  batch tile (wave id)
  int jt   = blockIdx.x;                            // 0..63 hidden tile
  int half = lane >> 4, ncol = lane & 15;
  int n = jt * 16 + ncol;                           // hidden unit index

  // --- one-time: persist this block's Wh^T slice into LDS (async copy) ---
  // LDS row r (0..63): gate g = r>>4, col c = r&15  <-  Wht row g*HID + jt*16 + c
  for (int idx = threadIdx.x; idx < 64 * 128; idx += 128) {
    int r = idx >> 7, chunk = idx & 127;            // 128 x 16B chunks per row
    int g = r >> 4, c = r & 15;
    unsigned lds_addr =
        (unsigned)(size_t)(bsl + r * LDSLD + chunk * 8);            // LDS byte addr
    const __bf16* gsrc =
        Wht + (size_t)(g * HID + jt * 16 + c) * HID + chunk * 8;
    // GLOBAL_LOAD_ASYNC_TO_LDS_B128 (GV mode): VDST = LDS addr, VADDR = 64b addr
    asm volatile("global_load_async_to_lds_b128 %0, %1, off"
                 :: "v"(lds_addr), "v"((unsigned long long)(size_t)gsrc)
                 : "memory");
  }
  asm volatile("s_wait_asynccnt 0x0" ::: "memory");
  __syncthreads();

  float* out_h = out + (size_t)SEQ * BATCH * HID;
  float* out_c = out_h + BATCH * HID;

  v8f cstate = v8f{};                               // cell tile in registers

  for (int t = 0; t < SEQ; ++t) {
    const __bf16* h_prev = hbf + (size_t)(t & 1) * BATCH * HID;
    __bf16*       h_next = hbf + (size_t)((t + 1) & 1) * BATCH * HID;

    if (t + 1 < SEQ)  // warm L2 for next step's gate slice
      __builtin_prefetch(gx + ((size_t)(t + 1) * BATCH + mt * 16 + ncol) * G4 + jt * 16, 0, 1);

    v8f acc[4] = {v8f{}, v8f{}, v8f{}, v8f{}};
    if (t > 0) {                                    // h_0 == 0: skip GEMM
      v16bf a_cur = load_a_bf16(h_prev, mt * 16, HID, 0, lane);
      v16bf b_cur[4];
#pragma unroll
      for (int g = 0; g < 4; ++g) b_cur[g] = load_b_lds(bsl, g * 16, 0, lane);
#pragma unroll 4
      for (int kt = 0; kt < HID; kt += 32) {
        int kn = (kt + 32) & (HID - 1);             // wrap: last prefetch harmless
        v16bf a_nxt = load_a_bf16(h_prev, mt * 16, HID, kn, lane);
        v16bf b_nxt[4];
#pragma unroll
        for (int g = 0; g < 4; ++g) b_nxt[g] = load_b_lds(bsl, g * 16, kn, lane);
#pragma unroll
        for (int g = 0; g < 4; ++g) acc[g] = wmma_bf16(a_cur, b_cur[g], acc[g]);
        a_cur = a_nxt;
#pragma unroll
        for (int g = 0; g < 4; ++g) b_cur[g] = b_nxt[g];
      }
    }

#pragma unroll
    for (int v = 0; v < 8; ++v) {
      int m = mt * 16 + v + 8 * half;               // batch index
      const float* g0 = gx + ((size_t)t * BATCH + m) * G4 + n;
      float ig = fast_sigmoid(g0[0]       + acc[0][v]);
      float fg = fast_sigmoid(g0[HID]     + acc[1][v]);
      float gg = fast_tanh   (g0[2 * HID] + acc[2][v]);
      float og = fast_sigmoid(g0[3 * HID] + acc[3][v]);
      float cv = fg * cstate[v] + ig * gg;
      cstate[v] = cv;
      float hv = og * fast_tanh(cv);
      out[(size_t)t * BATCH * HID + (size_t)m * HID + n] = hv;   // output[0][t]
      h_next[(size_t)m * HID + n] = (__bf16)hv;                  // next-step A
      if (t == SEQ - 1) {
        out_h[(size_t)m * HID + n] = hv;
        out_c[(size_t)m * HID + n] = cv;
      }
    }

    if (t + 1 < SEQ) grid_barrier(bar, bar + 1);    // publish h_t
  }
}

// ============================== launcher ==================================
extern "C" void kernel_launch(void* const* d_in, const int* in_sizes, int n_in,
                              void* d_out, int out_size, void* d_ws, size_t ws_size,
                              hipStream_t stream) {
  const float* x  = (const float*)d_in[0];
  const float* Wi = (const float*)d_in[1];
  const float* Wh = (const float*)d_in[2];
  const float* bi = (const float*)d_in[3];
  const float* bh = (const float*)d_in[4];
  float* out = (float*)d_out;

  char* ws = (char*)d_ws;
  __bf16*   Wit  = (__bf16*)(ws + WIT_OFF);
  __bf16*   Wht  = (__bf16*)(ws + WHT_OFF);
  float*    bias = (float*)(ws + BIAS_OFF);
  __bf16*   hbf  = (__bf16*)(ws + HBF_OFF);
  unsigned* bar  = (unsigned*)(ws + BAR_OFF);
  float*    gx   = (float*)(ws + GX_OFF);

  dim3 tb(32, 8);
  dim3 tg(INSZ / 32, G4 / 32);
  lstm_transpose_cvt<<<tg, tb, 0, stream>>>(Wi, Wit, INSZ, G4);
  lstm_transpose_cvt<<<tg, tb, 0, stream>>>(Wh, Wht, HID,  G4);

  lstm_prep<<<(BATCH * HID) / 256, 256, 0, stream>>>(bi, bh, bias, hbf, bar);

  // 2048 M-tiles x 64 wave-columns = 131072 waves = 16384 blocks
  lstm_xgemm<<<16384, 256, 0, stream>>>(x, Wit, bias, gx);

  lstm_recurrent<<<NBLK_PERSIST, 128, 0, stream>>>(gx, Wht, hbf, out, bar);
}
